// TiledSparseRead_63642825392623
// MI455X (gfx1250) — matrix operand order repaired and checked by reading, not compiled, not run on hardware
//
#include <hip/hip_runtime.h>
#include <hip/hip_bf16.h>
#include <stdint.h>

// Problem constants (from reference): D=4096, M=1024, B=64, N=2048
// T = 32, G = 64 groups, e = 16, tile_size = 128.
#define NROWS   2048
#define DDIM    4096
#define GGRP    64
#define EE      16
#define TSZ     128     // K dimension per group
#define BB      64
#define COLS_G  1024    // e*B columns per group
#define OUTROW  65536   // M*B floats per output row

typedef _Float16 v16h __attribute__((ext_vector_type(16)));
typedef _Float16 v8h  __attribute__((ext_vector_type(8)));
typedef float    v8f  __attribute__((ext_vector_type(8)));
typedef int      v4i  __attribute__((ext_vector_type(4)));

#define GLOBAL_AS __attribute__((address_space(1)))
#define LDS_AS    __attribute__((address_space(3)))

// ---------------------------------------------------------------------------
// Kernel 1: per-column inverse norms (f32) + V transposed to f16 Vt[g][c][t]
// ---------------------------------------------------------------------------
__global__ void prep_v_kernel(const float* __restrict__ V,
                              float* __restrict__ inv_norm,
                              _Float16* __restrict__ Vt) {
    const int slab = blockIdx.x;            // g*16 + ei, 1024 slabs
    const int b    = threadIdx.x;           // 0..63
    const float* vs = V + (size_t)slab * TSZ * BB;   // (t, b) slab
    float s = 0.f;
    #pragma unroll 4
    for (int t = 0; t < TSZ; ++t) {
        float v = vs[t * BB + b];           // coalesced across b
        s += v * v;
    }
    inv_norm[slab * BB + b] = rsqrtf(s);
    // transposed f16 copy: Vt[(g*1024 + ei*64 + b)][t]
    _Float16* dst = Vt + ((size_t)slab * BB + b) * TSZ;
    #pragma unroll 4
    for (int t = 0; t < TSZ; ++t)
        dst[t] = (_Float16)vs[t * BB + b];
}

// ---------------------------------------------------------------------------
// Kernel 2: gather x via tile_idx, convert to f16: Xh[g][n][t]
// ---------------------------------------------------------------------------
__global__ void prep_x_kernel(const float* __restrict__ x,
                              const int* __restrict__ tile_idx,
                              _Float16* __restrict__ Xh) {
    const int g = blockIdx.x;               // 0..63
    const int n = blockIdx.y;               // 0..2047
    const int t = threadIdx.x;              // 0..127
    const int idx = tile_idx[g * TSZ + t];
    Xh[((size_t)g * NROWS + n) * TSZ + t] = (_Float16)x[(size_t)n * DDIM + idx];
}

// ---------------------------------------------------------------------------
// Kernel 3: per-group GEMM with WMMA f16 -> f32, inv_norm scaling fused.
// Block = 256 threads (8 waves). Tile: 128 rows x 64 cols, K = 128.
// ---------------------------------------------------------------------------
#define APAD_H 136                          // padded LDS row stride in halves (272B)
#define OPAD_F 68                           // padded LDS out row stride in floats

__device__ __forceinline__ void stage16(const _Float16* __restrict__ src,
                                        _Float16* dst) {
#if __has_builtin(__builtin_amdgcn_global_load_async_to_lds_b128)
    __builtin_amdgcn_global_load_async_to_lds_b128(
        (GLOBAL_AS v4i*)(void*)src, (LDS_AS v4i*)(void*)dst, 0, 0);
#else
    *(v8h*)dst = *(const v8h*)src;
#endif
}

__device__ __forceinline__ void wait_stage() {
#if __has_builtin(__builtin_amdgcn_global_load_async_to_lds_b128)
#if __has_builtin(__builtin_amdgcn_s_wait_asynccnt)
    __builtin_amdgcn_s_wait_asynccnt(0);
#else
    asm volatile("s_wait_asynccnt 0" ::: "memory");
#endif
#endif
}

__global__ void __launch_bounds__(256)
gemm_wmma_kernel(const _Float16* __restrict__ Xh,
                 const _Float16* __restrict__ Vt,
                 const float* __restrict__ inv_norm,
                 float* __restrict__ out) {
    __shared__ alignas(16) char smem[(128 + 64) * APAD_H * sizeof(_Float16)]; // 52224 B
    _Float16* ldsA = (_Float16*)smem;                 // 128 x 136 halves
    _Float16* ldsB = ldsA + 128 * APAD_H;             // 64 x 136 halves
    float*    ldsO = (float*)smem;                    // aliased: 128 x 68 f32 (34816 B)

    const int ctile = blockIdx.x;                     // 0..15 -> c0 = ctile*64
    const int ntile = blockIdx.y;                     // 0..15 -> n0 = ntile*128
    const int g     = blockIdx.z;                     // 0..63
    const int c0    = ctile * 64;
    const int n0    = ntile * 128;

    const int tid  = threadIdx.x;
    const int wave = tid >> 5;                        // 0..7
    const int lane = tid & 31;
    const int wr   = wave >> 1;                       // row group 0..3 (32 rows)
    const int wc   = wave & 1;                        // col group 0..1 (32 cols)
    const int half = lane >> 4;                       // 0/1
    const int nn   = lane & 15;                       // matrix row (A) / col (B,C)

    // ---- stage A (128x128 halves) and B (64x128 halves) into LDS ----
    const _Float16* gA = Xh + ((size_t)g * NROWS + n0) * TSZ;
    const _Float16* gB = Vt + ((size_t)g * COLS_G + c0) * TSZ;
    {
        const int c  = tid & 15;                      // 16B chunk within row
        const int r0 = tid >> 4;                      // 0..15
        #pragma unroll
        for (int i = 0; i < 8; ++i) {                 // A: 128 rows
            const int r = r0 + i * 16;
            stage16(gA + r * TSZ + c * 8, ldsA + r * APAD_H + c * 8);
        }
        #pragma unroll
        for (int i = 0; i < 4; ++i) {                 // B: 64 rows
            const int r = r0 + i * 16;
            stage16(gB + r * TSZ + c * 8, ldsB + r * APAD_H + c * 8);
        }
    }
    wait_stage();
    __syncthreads();

    // ---- WMMA main: 2x2 tiles of 16x16 per wave, K = 4 x 32 ----
    v8f acc[2][2] = {};
    #pragma unroll
    for (int kk = 0; kk < 4; ++kk) {
        v16h a[2], b[2];
        #pragma unroll
        for (int rt = 0; rt < 2; ++rt) {
            const int ro = (wr * 32 + rt * 16 + nn) * APAD_H + kk * 32;
            *(v8h*)&a[rt]       = *(const v8h*)&ldsA[ro + half * 8];       // K: half*8..+7
            *((v8h*)&a[rt] + 1) = *(const v8h*)&ldsA[ro + 16 + half * 8];  // K: 16+half*8..
        }
        #pragma unroll
        for (int ct = 0; ct < 2; ++ct) {
            const int ro = (wc * 32 + ct * 16 + nn) * APAD_H + kk * 32;
            *(v8h*)&b[ct]       = *(const v8h*)&ldsB[ro + half * 8];       // K: half*8..+7
            *((v8h*)&b[ct] + 1) = *(const v8h*)&ldsB[ro + 16 + half * 8];  // K: 16+half*8..
        }
        #pragma unroll
        for (int rt = 0; rt < 2; ++rt)
            #pragma unroll
            for (int ct = 0; ct < 2; ++ct)
                acc[rt][ct] = __builtin_amdgcn_wmma_f32_16x16x32_f16(
                    false, a[rt], false, b[ct], (short)0, acc[rt][ct], false, false);
    }

    // ---- fuse inv_norm scaling (column-constant per lane) ----
    #pragma unroll
    for (int ct = 0; ct < 2; ++ct) {
        const float sc = inv_norm[g * COLS_G + c0 + wc * 32 + ct * 16 + nn];
        #pragma unroll
        for (int rt = 0; rt < 2; ++rt)
            #pragma unroll
            for (int r = 0; r < 8; ++r)
                acc[rt][ct][r] *= sc;
    }

    __syncthreads();   // done reading staged A/B; reuse LDS for output restage

    // ---- restage accumulators to LDS for contiguous 256B-row global stores ----
    #pragma unroll
    for (int rt = 0; rt < 2; ++rt)
        #pragma unroll
        for (int ct = 0; ct < 2; ++ct)
            #pragma unroll
            for (int r = 0; r < 8; ++r) {
                const int row = wr * 32 + rt * 16 + half * 8 + r;
                const int col = wc * 32 + ct * 16 + nn;
                ldsO[row * OPAD_F + col] = acc[rt][ct][r];
            }
    __syncthreads();

    // ---- coalesced stores: one 256B row per wave per iteration ----
    float* gO = out + (size_t)n0 * OUTROW + (size_t)g * COLS_G + c0;
    #pragma unroll
    for (int it = 0; it < 16; ++it) {
        const int row = wave + it * 8;
        const float2 v = *(const float2*)&ldsO[row * OPAD_F + lane * 2];
        *(float2*)(gO + (size_t)row * OUTROW + lane * 2) = v;
    }
}

// ---------------------------------------------------------------------------
// Host launcher
// ---------------------------------------------------------------------------
extern "C" void kernel_launch(void* const* d_in, const int* in_sizes, int n_in,
                              void* d_out, int out_size, void* d_ws, size_t ws_size,
                              hipStream_t stream) {
    const float* x   = (const float*)d_in[0];     // (2048, 4096)
    const float* V   = (const float*)d_in[1];     // (64, 16, 128, 64)
    const int*   tix = (const int*)d_in[2];       // (64, 128)

    // workspace layout
    char* ws = (char*)d_ws;
    float*    inv_norm = (float*)ws;                              // 65536 f32 = 256 KB
    _Float16* Vt = (_Float16*)(ws + 262144);                      // 64*1024*128 f16 = 16 MB
    _Float16* Xh = (_Float16*)(ws + 262144 + 16777216);           // 64*2048*128 f16 = 32 MB

    prep_v_kernel<<<GGRP * EE, BB, 0, stream>>>(V, inv_norm, Vt);
    prep_x_kernel<<<dim3(GGRP, NROWS), TSZ, 0, stream>>>(x, tix, Xh);
    gemm_wmma_kernel<<<dim3(16, 16, GGRP), 256, 0, stream>>>(Xh, Vt, inv_norm, (float*)d_out);
}
